// MemEffAttention_1254130450881
// MI455X (gfx1250) — matrix operand order repaired and checked by reading, not compiled, not run on hardware
//
#include <hip/hip_runtime.h>

// ---------------------------------------------------------------------------
// MI455X (gfx1250) memory-efficient attention:
//   bf16 WMMA (v_wmma_f32_16x16x32_bf16) everywhere, f32 accumulation.
//   Softmax runs in log2 domain: scale*log2(e) folded into Q projection.
// ---------------------------------------------------------------------------

typedef __attribute__((ext_vector_type(16))) __bf16 v16bf;
typedef __attribute__((ext_vector_type(8)))  __bf16 bf16x8;
typedef __attribute__((ext_vector_type(8)))  float  v8f;

#define DIM   1024
#define NH    16
#define HD    64
#define BB    4
#define TT    2048
#define MROWS (BB * TT)   // 8192

// hd^-0.5 * log2(e): folded into Q so S = log2-domain logits directly.
#define QSCALE (0.125f * 1.4426950408889634f)

__device__ __forceinline__ v16bf cat8(bf16x8 lo, bf16x8 hi) {
  return __builtin_shufflevector(lo, hi, 0,1,2,3,4,5,6,7,8,9,10,11,12,13,14,15);
}

__device__ __forceinline__ v8f vzero() {
  v8f z = {0.f,0.f,0.f,0.f,0.f,0.f,0.f,0.f};
  return z;
}

// max(a,b) via v_med3_f32 (med3(a,b,+inf) == max(a,b)); avoids the NaN
// canonicalization v_max ops clang inserts around opaque WMMA results.
__device__ __forceinline__ float fast_max(float a, float b) {
  return __builtin_amdgcn_fmed3f(a, b, __builtin_inff());
}

// WMMA A-matrix fragment, 16x32 bf16 (M x K), per ISA 7.12.2:
//   lane l holds row m = l&15; kbase = (l<16) ? 0 : 8
//   VGPR 0-3 : K = kbase .. kbase+7      (contiguous 16B)
//   VGPR 4-7 : K = kbase+16 .. kbase+23  (contiguous 16B)
__device__ __forceinline__ v16bf load_a_frag(const __bf16* p, int stride, int lane) {
  const int m  = lane & 15;
  const int kb = (lane < 16) ? 0 : 8;
  const __bf16* q = p + (size_t)m * stride + kb;
  bf16x8 lo = *(const bf16x8*)(q);
  bf16x8 hi = *(const bf16x8*)(q + 16);
  return cat8(lo, hi);
}

// WMMA B-matrix fragment, 32x16 bf16 (K x N):
//   lane l holds column n = l&15; kbase = (l<16) ? 0 : 8
//   VGPR 0-3 : K = kbase .. kbase+7; VGPR 4-7 : K = kbase+16 .. kbase+23.
// src[n*stride + k] = B[k,n] (i.e. src is the N x K matrix, row-major).
__device__ __forceinline__ v16bf load_b_frag(const __bf16* p, int stride, int lane) {
  const int n  = lane & 15;
  const int kb = (lane < 16) ? 0 : 8;
  const __bf16* q = p + (size_t)n * stride + kb;
  bf16x8 lo = *(const bf16x8*)(q);
  bf16x8 hi = *(const bf16x8*)(q + 16);
  return cat8(lo, hi);
}

__device__ __forceinline__ v8f wmma_bf16(v16bf a, v16bf b, v8f c) {
  return __builtin_amdgcn_wmma_f32_16x16x32_bf16(
      /*neg_a=*/false, a, /*neg_b=*/false, b,
      /*c_mod=*/(short)0, c, /*reuse_a=*/false, /*reuse_b=*/false);
}

// ---------------------------------------------------------------------------
// fp32 -> bf16 conversion (grid-stride)
// ---------------------------------------------------------------------------
__global__ void cvt_f32_bf16(const float* __restrict__ src,
                             __bf16* __restrict__ dst, int n) {
  int i = blockIdx.x * blockDim.x + threadIdx.x;
  const int stride = gridDim.x * blockDim.x;
  for (; i < n; i += stride) dst[i] = (__bf16)src[i];
}

// ---------------------------------------------------------------------------
// NT GEMM: C[MROWS,DIM] = A[MROWS,DIM] * W[DIM,DIM]^T, A/W bf16 row-major.
//   MODE 0: C bf16 row-major
//   MODE 1: C f32  row-major
//   MODE 2: C bf16 stored head-transposed: Vt[b, h, d, t]  (for V)
//   MODE 3: C bf16 row-major, scaled by QSCALE (for Q)
// Block = 256 threads (8 waves), block tile 128x128, wave tile 32x64.
// ---------------------------------------------------------------------------
template <int MODE>
__global__ __launch_bounds__(256)
void gemm_nt_bf16(const __bf16* __restrict__ A, const __bf16* __restrict__ W,
                  void* __restrict__ Cout) {
  const int lane = threadIdx.x & 31;
  const int w    = threadIdx.x >> 5;
  const int wm   = w & 3;        // 4 wave rows
  const int wn   = w >> 2;       // 2 wave cols
  const int row0 = blockIdx.x * 128 + wm * 32;
  const int col0 = blockIdx.y * 128 + wn * 64;

  v8f acc[2][4];
#pragma unroll
  for (int mt = 0; mt < 2; ++mt)
#pragma unroll
    for (int nt = 0; nt < 4; ++nt) acc[mt][nt] = vzero();

  for (int k0 = 0; k0 < DIM; k0 += 32) {
    v16bf af[2];
#pragma unroll
    for (int mt = 0; mt < 2; ++mt)
      af[mt] = load_a_frag(A + (size_t)(row0 + mt * 16) * DIM + k0, DIM, lane);
    v16bf bfrag[4];
#pragma unroll
    for (int nt = 0; nt < 4; ++nt)
      bfrag[nt] = load_b_frag(W + (size_t)(col0 + nt * 16) * DIM + k0, DIM, lane);
#pragma unroll
    for (int mt = 0; mt < 2; ++mt)
#pragma unroll
      for (int nt = 0; nt < 4; ++nt)
        acc[mt][nt] = wmma_bf16(af[mt], bfrag[nt], acc[mt][nt]);
  }

  // C/D layout: lane holds col n = lane&15; VGPR i holds row i + (lane<16?0:8).
  const int nlane = lane & 15;
  const int moff  = (lane < 16) ? 0 : 8;
#pragma unroll
  for (int mt = 0; mt < 2; ++mt)
#pragma unroll
    for (int nt = 0; nt < 4; ++nt)
#pragma unroll
      for (int i = 0; i < 8; ++i) {
        const int r = row0 + mt * 16 + i + moff;
        const int c = col0 + nt * 16 + nlane;
        const float v = acc[mt][nt][i];
        if (MODE == 0) {
          ((__bf16*)Cout)[(size_t)r * DIM + c] = (__bf16)v;
        } else if (MODE == 1) {
          ((float*)Cout)[(size_t)r * DIM + c] = v;
        } else if (MODE == 3) {
          ((__bf16*)Cout)[(size_t)r * DIM + c] = (__bf16)(v * QSCALE);
        } else {
          const int b = r >> 11, t = r & (TT - 1);       // TT = 2048
          const int h = c >> 6,  d = c & (HD - 1);       // HD = 64
          ((__bf16*)Cout)[(((size_t)(b * NH + h)) * HD + d) * TT + t] = (__bf16)v;
        }
      }
}

// ---------------------------------------------------------------------------
// Flash attention: one block (4 waves) per (b, h, 128-row q tile).
// Each wave owns 32 q rows. Q (pre-scaled by QSCALE) and K in [B*T, D] bf16;
// V transposed Vt[b,h,d,t]. Per 32-key tile: 8 WMMAs for S, log2-domain
// online softmax (per-lane partial row sums, deferred reduction), 8 WMMAs O.
// ---------------------------------------------------------------------------
__global__ __launch_bounds__(128)
void flash_attn(const __bf16* __restrict__ Q, const __bf16* __restrict__ Kb,
                const __bf16* __restrict__ Vt, __bf16* __restrict__ O) {
  __shared__ __align__(16) __bf16 Plds[4][32][32];   // 8 KB: P tile per wave

  const int lane  = threadIdx.x & 31;
  const int w     = threadIdx.x >> 5;
  const int qt    = blockIdx.x;                 // 0..T/128-1
  const int h     = blockIdx.y;                 // 0..NH-1
  const int b     = blockIdx.z;                 // 0..B-1
  const int hc    = h * HD;
  const int qrow0 = b * TT + qt * 128 + w * 32; // row into [B*T, D]

  // Q fragments: [mt = q-row subtile][kk = d half], resident all loop long.
  v16bf qf[2][2];
#pragma unroll
  for (int mt = 0; mt < 2; ++mt) {
    const __bf16* qp = Q + (size_t)(qrow0 + mt * 16) * DIM + hc;
    qf[mt][0] = load_a_frag(qp,      DIM, lane);
    qf[mt][1] = load_a_frag(qp + 32, DIM, lane);
  }

  v8f o_acc[2][4];
#pragma unroll
  for (int mt = 0; mt < 2; ++mt)
#pragma unroll
    for (int dt = 0; dt < 4; ++dt) o_acc[mt][dt] = vzero();

  // mrow: running log2-domain row max (uniform across the 16-lane half).
  // lrow: PER-LANE partial softmax denominator (this lane's 2 columns per
  //       tile); cross-lane sum deferred to the epilogue.
  float mrow[2][8], lrow[2][8];
#pragma unroll
  for (int mt = 0; mt < 2; ++mt)
#pragma unroll
    for (int i = 0; i < 8; ++i) { mrow[mt][i] = -1e30f; lrow[mt][i] = 0.f; }

  const __bf16* Vh = Vt + ((size_t)(b * NH + h)) * HD * TT;
  const int moff  = (lane < 16) ? 0 : 8;
  const int nlane = lane & 15;

  for (int j = 0; j < TT; j += 32) {
    // ---- prefetch next K/V tile (global_prefetch_b8), split across waves ----
    const int jn = j + 32;
    if (jn < TT) {
      if (w == 0)      __builtin_prefetch(Kb + (size_t)(b * TT + jn + lane) * DIM + hc, 0, 3);
      else if (w == 1) __builtin_prefetch(Vh + (size_t)lane * TT + jn, 0, 3);
      else if (w == 2) __builtin_prefetch(Vh + (size_t)(32 + lane) * TT + jn, 0, 3);
    }

    // ---- S = (Q*QSCALE) K^T : two 16x32 tiles per wave (32 q rows) ----
    v8f s[2][2];
#pragma unroll
    for (int mt = 0; mt < 2; ++mt) { s[mt][0] = vzero(); s[mt][1] = vzero(); }
#pragma unroll
    for (int nt = 0; nt < 2; ++nt) {
      const __bf16* kp = Kb + (size_t)(b * TT + j + nt * 16) * DIM + hc;
      v16bf bk0 = load_b_frag(kp,      DIM, lane);   // d 0..31
      v16bf bk1 = load_b_frag(kp + 32, DIM, lane);   // d 32..63
#pragma unroll
      for (int mt = 0; mt < 2; ++mt) {
        s[mt][nt] = wmma_bf16(qf[mt][0], bk0, s[mt][nt]);
        s[mt][nt] = wmma_bf16(qf[mt][1], bk1, s[mt][nt]);
      }
    }

    // ---- log2-domain online softmax ----
#pragma unroll
    for (int mt = 0; mt < 2; ++mt)
#pragma unroll
      for (int i = 0; i < 8; ++i) {
        const float s0 = s[mt][0][i];
        const float s1 = s[mt][1][i];
        float mx = fast_max(s0, s1);          // v_med3: no canonicalize ops
#pragma unroll
        for (int msk = 1; msk < 16; msk <<= 1)
          mx = fast_max(mx, __shfl_xor(mx, msk, 32));
        const float mnew  = fast_max(mrow[mt][i], mx);
        const float alpha = __builtin_amdgcn_exp2f(mrow[mt][i] - mnew);
        mrow[mt][i] = mnew;
        const float p0 = __builtin_amdgcn_exp2f(s0 - mnew);
        const float p1 = __builtin_amdgcn_exp2f(s1 - mnew);
        lrow[mt][i] = lrow[mt][i] * alpha + (p0 + p1);   // per-lane partial
#pragma unroll
        for (int dt = 0; dt < 4; ++dt) o_acc[mt][dt][i] *= alpha;
        // P -> LDS row-major (C layout -> A-frag layout transpose)
        const int rr = mt * 16 + i + moff;
        Plds[w][rr][nlane]      = (__bf16)p0;
        Plds[w][rr][nlane + 16] = (__bf16)p1;
      }

    asm volatile("s_wait_dscnt 0" ::: "memory");   // own-wave P visibility

    // ---- O += P * V (contraction over 32 kv rows; vf reused across mt) ----
    v16bf pf[2];
    pf[0] = load_a_frag(&Plds[w][0][0],  32, lane);
    pf[1] = load_a_frag(&Plds[w][16][0], 32, lane);
#pragma unroll
    for (int dt = 0; dt < 4; ++dt) {
      v16bf vf = load_b_frag(Vh + (size_t)(dt * 16) * TT + j, TT, lane);
      o_acc[0][dt] = wmma_bf16(pf[0], vf, o_acc[0][dt]);
      o_acc[1][dt] = wmma_bf16(pf[1], vf, o_acc[1][dt]);
    }
  }

  // ---- epilogue: reduce per-lane l across the 16-lane half, O /= l ----
#pragma unroll
  for (int mt = 0; mt < 2; ++mt)
#pragma unroll
    for (int i = 0; i < 8; ++i) {
      float rs = lrow[mt][i];
#pragma unroll
      for (int msk = 1; msk < 16; msk <<= 1)
        rs += __shfl_xor(rs, msk, 32);
      const float inv = 1.0f / rs;
      const size_t r = (size_t)(qrow0 + mt * 16 + i + moff);
#pragma unroll
      for (int dt = 0; dt < 4; ++dt)
        O[r * DIM + hc + dt * 16 + nlane] = (__bf16)(o_acc[mt][dt][i] * inv);
    }
}

// ---------------------------------------------------------------------------
// Host launcher
// ---------------------------------------------------------------------------
extern "C" void kernel_launch(void* const* d_in, const int* in_sizes, int n_in,
                              void* d_out, int out_size, void* d_ws, size_t ws_size,
                              hipStream_t stream) {
  (void)in_sizes; (void)n_in; (void)out_size; (void)ws_size;

  const float* x  = (const float*)d_in[0];
  const float* Wq = (const float*)d_in[1];
  const float* Wk = (const float*)d_in[2];
  const float* Wv = (const float*)d_in[3];
  const float* Wo = (const float*)d_in[4];
  float* out = (float*)d_out;

  const size_t SZ_X = (size_t)MROWS * DIM;  // 8388608
  const size_t SZ_W = (size_t)DIM * DIM;    // 1048576

  __bf16* xb  = (__bf16*)d_ws;
  __bf16* wqb = xb  + SZ_X;
  __bf16* wkb = wqb + SZ_W;
  __bf16* wvb = wkb + SZ_W;
  __bf16* wob = wvb + SZ_W;
  __bf16* qb  = wob + SZ_W;
  __bf16* kb  = qb  + SZ_X;
  __bf16* vt  = kb  + SZ_X;
  __bf16* ob  = vt  + SZ_X;   // total ~92.3 MB of workspace

  // 1) fp32 -> bf16
  cvt_f32_bf16<<<512, 256, 0, stream>>>(x,  xb,  (int)SZ_X);
  cvt_f32_bf16<<<256, 256, 0, stream>>>(Wq, wqb, (int)SZ_W);
  cvt_f32_bf16<<<256, 256, 0, stream>>>(Wk, wkb, (int)SZ_W);
  cvt_f32_bf16<<<256, 256, 0, stream>>>(Wv, wvb, (int)SZ_W);
  cvt_f32_bf16<<<256, 256, 0, stream>>>(Wo, wob, (int)SZ_W);

  // 2) projections: Q = (x Wq^T)*QSCALE, K = x Wk^T, V -> Vt[b,h,d,t]
  dim3 ggrid(MROWS / 128, DIM / 128);
  gemm_nt_bf16<3><<<ggrid, 256, 0, stream>>>(xb, wqb, qb);
  gemm_nt_bf16<0><<<ggrid, 256, 0, stream>>>(xb, wkb, kb);
  gemm_nt_bf16<2><<<ggrid, 256, 0, stream>>>(xb, wvb, vt);

  // 3) attention
  flash_attn<<<dim3(TT / 128, NH, BB), 128, 0, stream>>>(qb, kb, vt, ob);

  // 4) output projection straight to f32 d_out
  gemm_nt_bf16<1><<<ggrid, 256, 0, stream>>>(ob, wob, out);
}
